// MultiHeadSelfAttention_48043504173327
// MI455X (gfx1250) — matrix operand order repaired and checked by reading, not compiled, not run on hardware
//
#include <hip/hip_runtime.h>
#include <cstdint>
#include <cstddef>

// ---------------------------------------------------------------------------
// MHA forward for MI455X (gfx1250, wave32, WMMA).
// GEMMs: C = A * B^T, A/B row-major [*,K] bf16, f32 accumulate via
// v_wmma_f32_16x16x32_bf16. 128x128 block tile, 8 waves (2x4), each wave a
// 64x32 sub-tile = 4x2 WMMA accumulators. K staged through double-buffered
// LDS (+8 bf16 row pad) using gfx1250 GLOBAL_LOAD_ASYNC_TO_LDS_B128
// (ASYNCcnt-tracked async data mover), fallback: b128 load + ds_store.
// ---------------------------------------------------------------------------

typedef __attribute__((ext_vector_type(16))) __bf16   v16bf;
typedef __attribute__((ext_vector_type(8)))  float    v8f;
typedef __attribute__((ext_vector_type(4)))  uint32_t u32x4;
typedef __attribute__((ext_vector_type(2)))  uint32_t u32x2;

// builtin's exact parameter types: vector_size(16) int in AS1 / AS3
typedef int v4i_vs __attribute__((vector_size(16)));
typedef __attribute__((address_space(1))) v4i_vs as1_v4i;
typedef __attribute__((address_space(3))) v4i_vs as3_v4i;

#define D_MODEL   1024
#define SEQ       2048
#define BATCH     4
#define NUM_HEADS 16
#define HEAD_DIM  64
#define MTOT      (BATCH * SEQ)   // 8192 rows
#define LDK       40              // 32 K + 8 bf16 pad (keeps 16B alignment)

#if __has_builtin(__builtin_amdgcn_global_load_async_to_lds_b128)
#define ASYNC_CP 1
#else
#define ASYNC_CP 0
#endif

__device__ __forceinline__ void async_wait_all() {
#if ASYNC_CP
#if __has_builtin(__builtin_amdgcn_s_wait_asynccnt)
  __builtin_amdgcn_s_wait_asynccnt(0);
#else
  asm volatile("s_wait_asynccnt 0x0" ::: "memory");
#endif
#endif
}

// copy 16B global -> LDS (async data mover when available)
__device__ __forceinline__ void cp_b128(const __bf16* gsrc, __bf16* lds) {
#if ASYNC_CP
  __builtin_amdgcn_global_load_async_to_lds_b128(
      (as1_v4i*)gsrc, (as3_v4i*)lds, 0, 0);
#else
  *(u32x4*)lds = *(const u32x4*)gsrc;
#endif
}

// ---------------- f32 -> bf16 convert (vectorized x4) ----------------
__global__ void cvt_f32_bf16(const float* __restrict__ in,
                             __bf16* __restrict__ out, int n4) {
  for (int i = blockIdx.x * blockDim.x + threadIdx.x; i < n4;
       i += gridDim.x * blockDim.x) {
    const float4 f = ((const float4*)in)[i];
    __bf16 t[4] = {(__bf16)f.x, (__bf16)f.y, (__bf16)f.z, (__bf16)f.w};
    *(u32x2*)&out[(size_t)i * 4] = *(u32x2*)t;
  }
}

// ---------------- generic bf16 WMMA GEMM: C = A * B^T ----------------
// mode 0: outF[row*ldc+col] = acc*scale (+bias)          (f32)
// mode 1: outB[row*ldc+col] = bf16(acc*scale (+bias))    (bf16 row-major)
// mode 2: V-transpose epilogue: row->(b,s), col->(h,d), write vt[b,h,d,s]
// batched: gridDim.z = B*NUM_HEADS; offsets A/B by head slice, outF by z.
__global__ __launch_bounds__(256)
void gemm_abt_bf16(const __bf16* __restrict__ A, int lda,
                   const __bf16* __restrict__ B, int ldb,
                   const float* __restrict__ bias,
                   float scale, int mode, int kdim,
                   float* __restrict__ outF, __bf16* __restrict__ outB,
                   int ldc, int batched) {
  __shared__ __bf16 ldsA[2][128 * LDK];
  __shared__ __bf16 ldsB[2][128 * LDK];

  const int tid  = threadIdx.x;
  const int lane = tid & 31;
  const int wid  = tid >> 5;
  const int wm   = (wid & 1) * 64;   // wave row offset in tile
  const int wn   = (wid >> 1) * 32;  // wave col offset in tile
  const int m0   = blockIdx.y * 128;
  const int n0   = blockIdx.x * 128;

  if (batched) {
    const int z = blockIdx.z;
    const int b = z >> 4, h = z & 15;
    A    += (size_t)b * SEQ * lda + (size_t)h * HEAD_DIM;
    B    += (size_t)b * SEQ * ldb + (size_t)h * HEAD_DIM;
    outF += (size_t)z * SEQ * SEQ;
  }

  // Staging addressing: 2 x 16B chunks per thread for each of A and B.
  const int rS = tid >> 2;             // 0..63
  const int cS = (tid & 3) << 3;       // 0,8,16,24
  const __bf16* aG0 = A + (size_t)(m0 + rS) * lda + cS;
  const __bf16* aG1 = A + (size_t)(m0 + rS + 64) * lda + cS;
  const __bf16* bG0 = B + (size_t)(n0 + rS) * ldb + cS;
  const __bf16* bG1 = B + (size_t)(n0 + rS + 64) * ldb + cS;
  const int lOff0 = rS * LDK + cS;
  const int lOff1 = (rS + 64) * LDK + cS;

  v8f acc[4][2];
  const v8f vzero = {};
  for (int i = 0; i < 4; ++i)
    for (int j = 0; j < 2; ++j) acc[i][j] = vzero;

  const int khalf = (lane >> 4) << 3;  // A frag K group (ISA 16-bit A layout)
  const int kst   = (lane >> 4) << 4;  // unused; B frag uses khalf like A
  const int mrow  = lane & 15;

  // prologue: stage first K tile into buffer 0
  cp_b128(aG0, &ldsA[0][lOff0]);
  cp_b128(aG1, &ldsA[0][lOff1]);
  cp_b128(bG0, &ldsB[0][lOff0]);
  cp_b128(bG1, &ldsB[0][lOff1]);

  int buf = 0;
  for (int k0 = 0; k0 < kdim; k0 += 32) {
    async_wait_all();
    __syncthreads();                    // staged tile visible to all waves
    const int kn = k0 + 32;
    if (kn < kdim) {                    // stage next tile into other buffer
      cp_b128(aG0 + kn, &ldsA[buf ^ 1][lOff0]);
      cp_b128(aG1 + kn, &ldsA[buf ^ 1][lOff1]);
      cp_b128(bG0 + kn, &ldsB[buf ^ 1][lOff0]);
      cp_b128(bG1 + kn, &ldsB[buf ^ 1][lOff1]);
    }

    v16bf afrag[4], bfrag[2];
    for (int tm = 0; tm < 4; ++tm) {
      const int row = wm + tm * 16 + mrow;
      u32x4* p = (u32x4*)&afrag[tm];
      p[0] = *(const u32x4*)&ldsA[buf][row * LDK + khalf];
      p[1] = *(const u32x4*)&ldsA[buf][row * LDK + 16 + khalf];
    }
    for (int tn = 0; tn < 2; ++tn) {
      const int col = wn + tn * 16 + mrow;
      u32x4* p = (u32x4*)&bfrag[tn];
      p[0] = *(const u32x4*)&ldsB[buf][col * LDK + khalf];
      p[1] = *(const u32x4*)&ldsB[buf][col * LDK + 16 + khalf];
    }
    for (int tm = 0; tm < 4; ++tm)
      for (int tn = 0; tn < 2; ++tn)
        acc[tm][tn] = __builtin_amdgcn_wmma_f32_16x16x32_bf16(
            false, afrag[tm], false, bfrag[tn], (short)0, acc[tm][tn],
            false, false);
    buf ^= 1;
  }

  // Epilogue. C layout: VGPR i, lane l -> row = i + (l>=16)*8, col = l&15.
  const int rh = (lane >> 4) << 3;
  const int cl = lane & 15;
  for (int tm = 0; tm < 4; ++tm)
    for (int tn = 0; tn < 2; ++tn)
      for (int i = 0; i < 8; ++i) {
        const int grow = m0 + wm + tm * 16 + i + rh;
        const int gcol = n0 + wn + tn * 16 + cl;
        float v = acc[tm][tn][i] * scale;
        if (bias) v += bias[gcol];
        if (mode == 0) {
          outF[(size_t)grow * ldc + gcol] = v;
        } else if (mode == 1) {
          outB[(size_t)grow * ldc + gcol] = (__bf16)v;
        } else {  // V^T per head: [b, h, d, s]
          const int b = grow >> 11, s = grow & 2047;
          const int h = gcol >> 6,  d = gcol & 63;
          outB[(((size_t)(b * NUM_HEADS + h) * HEAD_DIM + d) << 11) + s] =
              (__bf16)v;
        }
      }
}

// ---------------- row softmax over SEQ=2048, one block per row --------------
__global__ __launch_bounds__(256)
void softmax_kernel(float* __restrict__ attn) {
  float* p = attn + (size_t)blockIdx.x * SEQ;
  const int tid = threadIdx.x;
  float vals[8];
  float mx = -3.402823466e38f;
  for (int j = 0; j < 8; ++j) {
    vals[j] = p[tid + (j << 8)];
    mx = fmaxf(mx, vals[j]);
  }
  __shared__ float red[256];
  red[tid] = mx;
  __syncthreads();
  for (int s = 128; s > 0; s >>= 1) {
    if (tid < s) red[tid] = fmaxf(red[tid], red[tid + s]);
    __syncthreads();
  }
  mx = red[0];
  __syncthreads();
  float sum = 0.f;
  for (int j = 0; j < 8; ++j) {
    vals[j] = __expf(vals[j] - mx);
    sum += vals[j];
  }
  red[tid] = sum;
  __syncthreads();
  for (int s = 128; s > 0; s >>= 1) {
    if (tid < s) red[tid] += red[tid + s];
    __syncthreads();
  }
  const float inv = 1.f / red[0];
  for (int j = 0; j < 8; ++j) p[tid + (j << 8)] = vals[j] * inv;
}

// ---------------- context = attn(f32) * V^T(bf16): per (b,h) ---------------
// M=2048, N=64, K=2048. 128x64 block tile, 8 waves 4x2, each 32x32 (2x2 WMMA).
// attn converted f32->bf16 on the LDS staging path; V^T tile async-copied.
__global__ __launch_bounds__(256)
void context_kernel(const float* __restrict__ attn,
                    const __bf16* __restrict__ vt,
                    __bf16* __restrict__ ctx) {
  __shared__ __bf16 ldsA[2][128 * LDK];
  __shared__ __bf16 ldsB[2][64 * LDK];

  const int tid = threadIdx.x, lane = tid & 31, wid = tid >> 5;
  const int wm = (wid >> 1) * 32;  // 0..96
  const int wn = (wid & 1) * 32;   // 0 or 32
  const int m0 = blockIdx.x * 128;
  const int z  = blockIdx.z;       // b*16 + h
  const int b  = z >> 4, h = z & 15;
  const float*  Ap = attn + (size_t)z * SEQ * SEQ;
  const __bf16* Bp = vt   + (size_t)z * HEAD_DIM * SEQ;

  // A staging: 4 float4 chunks per thread (128x32 f32 -> bf16)
  const int rA = tid >> 3;             // 0..31
  const int cA = (tid & 7) << 2;       // 0..28 step 4
  const float* aG[4];
  int aL[4];
  for (int j = 0; j < 4; ++j) {
    aG[j] = Ap + (size_t)(m0 + rA + 32 * j) * SEQ + cA;
    aL[j] = (rA + 32 * j) * LDK + cA;
  }
  // B staging: 1 x 16B chunk per thread (64x32 bf16)
  const int rB = tid >> 2, cB = (tid & 3) << 3;
  const __bf16* bG = Bp + (size_t)rB * SEQ + cB;
  const int bL = rB * LDK + cB;

  v8f acc[2][2];
  const v8f vzero = {};
  for (int i = 0; i < 2; ++i)
    for (int j = 0; j < 2; ++j) acc[i][j] = vzero;

  const int khalf = (lane >> 4) << 3;
  const int kst   = (lane >> 4) << 4;
  const int mrow  = lane & 15;

  // prologue stage into buffer 0
  for (int j = 0; j < 4; ++j) {
    const float4 f = *(const float4*)aG[j];
    __bf16 t[4] = {(__bf16)f.x, (__bf16)f.y, (__bf16)f.z, (__bf16)f.w};
    *(u32x2*)&ldsA[0][aL[j]] = *(u32x2*)t;
  }
  cp_b128(bG, &ldsB[0][bL]);

  int buf = 0;
  for (int k0 = 0; k0 < SEQ; k0 += 32) {
    async_wait_all();
    __syncthreads();
    const int kn = k0 + 32;
    if (kn < SEQ) {
      for (int j = 0; j < 4; ++j) {
        const float4 f = *(const float4*)(aG[j] + kn);
        __bf16 t[4] = {(__bf16)f.x, (__bf16)f.y, (__bf16)f.z, (__bf16)f.w};
        *(u32x2*)&ldsA[buf ^ 1][aL[j]] = *(u32x2*)t;
      }
      cp_b128(bG + kn, &ldsB[buf ^ 1][bL]);
    }

    v16bf afrag[2], bfrag[2];
    for (int tm = 0; tm < 2; ++tm) {
      const int row = wm + tm * 16 + mrow;
      u32x4* p = (u32x4*)&afrag[tm];
      p[0] = *(const u32x4*)&ldsA[buf][row * LDK + khalf];
      p[1] = *(const u32x4*)&ldsA[buf][row * LDK + 16 + khalf];
    }
    for (int tn = 0; tn < 2; ++tn) {
      const int col = wn + tn * 16 + mrow;
      u32x4* p = (u32x4*)&bfrag[tn];
      p[0] = *(const u32x4*)&ldsB[buf][col * LDK + khalf];
      p[1] = *(const u32x4*)&ldsB[buf][col * LDK + 16 + khalf];
    }
    for (int tm = 0; tm < 2; ++tm)
      for (int tn = 0; tn < 2; ++tn)
        acc[tm][tn] = __builtin_amdgcn_wmma_f32_16x16x32_bf16(
            false, afrag[tm], false, bfrag[tn], (short)0, acc[tm][tn],
            false, false);
    buf ^= 1;
  }

  const int rh = (lane >> 4) << 3;
  const int cl = lane & 15;
  for (int tm = 0; tm < 2; ++tm)
    for (int tn = 0; tn < 2; ++tn)
      for (int i = 0; i < 8; ++i) {
        const int s = m0 + wm + tm * 16 + i + rh;  // sequence position
        const int d = wn + tn * 16 + cl;           // head dim 0..63
        ctx[(size_t)(b * SEQ + s) * D_MODEL + h * HEAD_DIM + d] =
            (__bf16)acc[tm][tn][i];
      }
}

// ---------------------------------------------------------------------------
extern "C" void kernel_launch(void* const* d_in, const int* in_sizes, int n_in,
                              void* d_out, int out_size, void* d_ws,
                              size_t ws_size, hipStream_t stream) {
  (void)in_sizes; (void)n_in; (void)out_size; (void)ws_size;
  const float* x  = (const float*)d_in[0];
  const float* wq = (const float*)d_in[1];
  const float* bq = (const float*)d_in[2];
  const float* wk = (const float*)d_in[3];
  const float* bk = (const float*)d_in[4];
  const float* wv = (const float*)d_in[5];
  const float* bv = (const float*)d_in[6];
  const float* wo = (const float*)d_in[7];
  const float* bo = (const float*)d_in[8];

  float* out  = (float*)d_out;                         // [4,2048,1024]
  float* attn = out + (size_t)MTOT * D_MODEL;          // [4,16,2048,2048]

  // Workspace layout (bytes): xbf 16M | qbf 16M | kbf 16M | vtbf 16M | 4x wbf
  char* ws = (char*)d_ws;
  const size_t NX = (size_t)MTOT * D_MODEL;      // 8,388,608
  const size_t NW = (size_t)D_MODEL * D_MODEL;   // 1,048,576
  __bf16* xbf   = (__bf16*)(ws);
  __bf16* qbf   = (__bf16*)(ws + 2 * NX);
  __bf16* kbf   = (__bf16*)(ws + 4 * NX);
  __bf16* vtbf  = (__bf16*)(ws + 6 * NX);
  __bf16* wqbf  = (__bf16*)(ws + 8 * NX);
  __bf16* wkbf  = (__bf16*)(ws + 8 * NX + 2 * NW);
  __bf16* wvbf  = (__bf16*)(ws + 8 * NX + 4 * NW);
  __bf16* wobf  = (__bf16*)(ws + 8 * NX + 6 * NW);
  __bf16* ctxbf = xbf;  // x no longer needed after QKV -> reuse

  // 1) converts (x4 vectorized)
  cvt_f32_bf16<<<1024, 256, 0, stream>>>(x, xbf, (int)(NX / 4));
  cvt_f32_bf16<<<256, 256, 0, stream>>>(wq, wqbf, (int)(NW / 4));
  cvt_f32_bf16<<<256, 256, 0, stream>>>(wk, wkbf, (int)(NW / 4));
  cvt_f32_bf16<<<256, 256, 0, stream>>>(wv, wvbf, (int)(NW / 4));
  cvt_f32_bf16<<<256, 256, 0, stream>>>(wo, wobf, (int)(NW / 4));

  const dim3 blk(256);
  // 2) QKV projections: [8192,1024] x [1024,1024]^T
  gemm_abt_bf16<<<dim3(8, 64, 1), blk, 0, stream>>>(
      xbf, D_MODEL, wqbf, D_MODEL, bq, 1.f, 1, D_MODEL, nullptr, qbf,
      D_MODEL, 0);
  gemm_abt_bf16<<<dim3(8, 64, 1), blk, 0, stream>>>(
      xbf, D_MODEL, wkbf, D_MODEL, bk, 1.f, 1, D_MODEL, nullptr, kbf,
      D_MODEL, 0);
  gemm_abt_bf16<<<dim3(8, 64, 1), blk, 0, stream>>>(
      xbf, D_MODEL, wvbf, D_MODEL, bv, 1.f, 2, D_MODEL, nullptr, vtbf,
      D_MODEL, 0);
  // 3) scores = Q K^T / 8 per (b,h): batched over z = 64
  gemm_abt_bf16<<<dim3(16, 16, BATCH * NUM_HEADS), blk, 0, stream>>>(
      qbf, D_MODEL, kbf, D_MODEL, nullptr, 0.125f, 0, HEAD_DIM, attn, nullptr,
      SEQ, 1);
  // 4) softmax in place on d_out attn region
  softmax_kernel<<<BATCH * NUM_HEADS * SEQ, 256, 0, stream>>>(attn);
  // 5) context = attn * V^T -> ctx bf16 [b,s,h*64+d]
  context_kernel<<<dim3(16, 1, BATCH * NUM_HEADS), blk, 0, stream>>>(
      attn, vtbf, ctxbf);
  // 6) output projection (f32 out + bias)
  gemm_abt_bf16<<<dim3(8, 64, 1), blk, 0, stream>>>(
      ctxbf, D_MODEL, wobf, D_MODEL, bo, 1.f, 0, D_MODEL, out, nullptr,
      D_MODEL, 0);
}
